// GAT_4827543240906
// MI455X (gfx1250) — matrix operand order repaired and checked by reading, not compiled, not run on hardware
//
#include <hip/hip_runtime.h>
#include <hip/hip_bf16.h>

typedef __attribute__((ext_vector_type(16))) __bf16 v16bf;
typedef __attribute__((ext_vector_type(8)))  float  v8f;

#define HEADS1 8
#define HID1   32
#define D1     256   // HEADS1*HID1
#define INCH   512
#define NCLS   40
#define NCOL2  64    // NCLS padded to 4 col-tiles
#define NEG_SLOPE 0.2f

// ---------------- atomics ----------------
__device__ __forceinline__ void atomicAddF(float* a, float v) {
  __hip_atomic_fetch_add(a, v, __ATOMIC_RELAXED, __HIP_MEMORY_SCOPE_AGENT);
}
__device__ __forceinline__ void atomicMaxF(float* addr, float v) {
  // monotonic int mapping: signed max for >=0, unsigned min for <0
  if (v >= 0.f) atomicMax((int*)addr, __float_as_int(v));
  else          atomicMin((unsigned int*)addr, (unsigned int)__float_as_int(v));
}

// ---------------- fills ----------------
__global__ void fill_f32(float* __restrict__ p, float v, long long n) {
  long long i = (long long)blockIdx.x * blockDim.x + threadIdx.x;
  if (i < n) p[i] = v;
}
__global__ void init_bias(float* __restrict__ out, const float* __restrict__ b,
                          long long n, int C) {
  long long i = (long long)blockIdx.x * blockDim.x + threadIdx.x;
  if (i < n) out[i] = b[i % C];
}

// -------- pack A (f32 [M,ldA]) into bf16 WMMA A-fragment order --------
// layout: frag[((rt*KS + ks)*32 + lane)*16 + i]
// A elem: M = rt*16 + lane%16 ; K = ks*32 + (i<8 ? i : i+8) + 8*(lane/16)
__global__ void conv_a_frag(const float* __restrict__ A, const float* __restrict__ bias,
                            int useElu, int M, int ldA, int KS,
                            __bf16* __restrict__ frag, long long total) {
  long long idx = (long long)blockIdx.x * blockDim.x + threadIdx.x;
  if (idx >= total) return;
  int i  = (int)(idx & 15);
  int l  = (int)((idx >> 4) & 31);
  long long rest = idx >> 9;
  int ks = (int)(rest % KS);
  long long rt = rest / KS;
  int m  = (int)(rt * 16 + (l & 15));
  int kd = ks * 32 + (i < 8 ? i : i + 8) + ((l >> 4) << 3);
  float v = 0.f;
  if (m < M) {
    v = A[(size_t)m * ldA + kd];
    if (bias)   v += bias[kd];
    if (useElu) v = v > 0.f ? v : (__expf(v) - 1.f);
  }
  frag[idx] = (__bf16)v;
}

// -------- pack B (f32 [K, realCols]) into bf16 WMMA B-fragment order --------
// layout: frag[((ct*KS + ks)*32 + lane)*16 + i]
// B elem: N = ct*16 + lane%16 ; K = ks*32 + (i<8 ? i : i+8) + 8*(lane/16)
__global__ void conv_b_frag(const float* __restrict__ W, int K, int realCols,
                            int KS, __bf16* __restrict__ frag, int total) {
  int idx = blockIdx.x * blockDim.x + threadIdx.x;
  if (idx >= total) return;
  int i  = idx & 15;
  int l  = (idx >> 4) & 31;
  int rest = idx >> 9;
  int ks = rest % KS;
  int ct = rest / KS;
  int col = ct * 16 + (l & 15);
  int kd  = ks * 32 + (i < 8 ? i : i + 8) + ((l >> 4) << 3);
  float v = (col < realCols && kd < K) ? W[(size_t)kd * realCols + col] : 0.f;
  frag[idx] = (__bf16)v;
}

// -------- WMMA GEMM: C[M,Ncol] = A * B, bf16 in / f32 out --------
// 8 waves per block; each wave owns one 16-row tile x CT 16-col tiles,
// reusing the A fragment from registers across the CT column tiles.
template <int CT>
__global__ void gemm_wmma_bf16(const __bf16* __restrict__ Afrag,
                               const __bf16* __restrict__ Bfrag,
                               float* __restrict__ C,
                               int Mtiles, int Ntiles, int KS, int Ncol, int M) {
  const int NG = Ntiles / CT;               // col-tile groups per row tile
  int wave = threadIdx.x >> 5;
  int lane = threadIdx.x & 31;
  int t = blockIdx.x * 8 + wave;
  if (t >= Mtiles * NG) return;
  int rt  = t / NG;
  int ct0 = (t % NG) * CT;

  v8f acc[CT];
#pragma unroll
  for (int j = 0; j < CT; ++j) acc[j] = (v8f){};

  const __bf16* aptr = Afrag + ((size_t)(rt * KS) * 32 + lane) * 16;
  const __bf16* bbase = Bfrag + ((size_t)(ct0 * KS) * 32 + lane) * 16;

  for (int ks = 0; ks < KS; ++ks) {
    v16bf a = *(const v16bf*)aptr;
#pragma unroll
    for (int j = 0; j < CT; ++j) {
      v16bf b = *(const v16bf*)(bbase + ((size_t)j * KS + ks) * 512);
      acc[j] = __builtin_amdgcn_wmma_f32_16x16x32_bf16(false, a, false, b,
                                                       (short)0, acc[j], false, false);
    }
    aptr += 512;  // 32 lanes * 16 elems
  }

  int row0 = rt * 16 + ((lane >> 4) << 3);
  int colL = lane & 15;
#pragma unroll
  for (int j = 0; j < CT; ++j) {
    int col = (ct0 + j) * 16 + colL;
#pragma unroll
    for (int r = 0; r < 8; ++r) {
      int row = row0 + r;
      if (row < M) C[(size_t)row * Ncol + col] = acc[j][r];
    }
  }
}

// -------- attention logits, 8 heads x 32 ch --------
__global__ void attn_logits8(const float* __restrict__ h, const float* __restrict__ asrc,
                             const float* __restrict__ adst, float* __restrict__ als,
                             float* __restrict__ ald, int N) {
  int idx = blockIdx.x * blockDim.x + threadIdx.x;  // n*8 + head
  if (idx >= N * HEADS1) return;
  int hh = idx & 7, n = idx >> 3;
  const float* hp = h + (size_t)n * D1 + hh * HID1;
  const float* as = asrc + hh * HID1;
  const float* ad = adst + hh * HID1;
  float s = 0.f, d = 0.f;
#pragma unroll
  for (int c = 0; c < HID1; ++c) { float v = hp[c]; s += v * as[c]; d += v * ad[c]; }
  als[idx] = s; ald[idx] = d;
}

// -------- attention logits, 1 head x C ch (ldh stride) --------
__global__ void attn_logits1(const float* __restrict__ h, const float* __restrict__ asrc,
                             const float* __restrict__ adst, float* __restrict__ als,
                             float* __restrict__ ald, int N, int C, int ldh) {
  int n = blockIdx.x * blockDim.x + threadIdx.x;
  if (n >= N) return;
  const float* hp = h + (size_t)n * ldh;
  float s = 0.f, d = 0.f;
  for (int c = 0; c < C; ++c) { float v = hp[c]; s += v * asrc[c]; d += v * adst[c]; }
  als[n] = s; ald[n] = d;
}

// -------- edge pass 1: leaky-relu logit, segment max --------
__global__ void edge_pass1(const int* __restrict__ ei, int E, int EE, int H,
                           const float* __restrict__ als, const float* __restrict__ ald,
                           float* __restrict__ p, float* __restrict__ m) {
  long long idx = (long long)blockIdx.x * blockDim.x + threadIdx.x;
  if (idx >= (long long)EE * H) return;
  int hh = (int)(idx % H);
  long long e = idx / H;
  int s, d;
  if (e < E) { s = ei[e]; d = ei[(size_t)E + e]; } else { s = d = (int)(e - E); }
  float ev = als[(size_t)s * H + hh] + ald[(size_t)d * H + hh];
  ev = ev > 0.f ? ev : NEG_SLOPE * ev;
  p[idx] = ev;
  atomicMaxF(&m[(size_t)d * H + hh], ev);
}

// -------- edge pass 2: exp(e - max), segment sum --------
__global__ void edge_pass2(const int* __restrict__ ei, int E, int EE, int H,
                           const float* __restrict__ m, float* __restrict__ p,
                           float* __restrict__ den) {
  long long idx = (long long)blockIdx.x * blockDim.x + threadIdx.x;
  if (idx >= (long long)EE * H) return;
  int hh = (int)(idx % H);
  long long e = idx / H;
  int d;
  if (e < E) { d = ei[(size_t)E + e]; } else { d = (int)(e - E); }
  float pv = __expf(p[idx] - m[(size_t)d * H + hh]);
  p[idx] = pv;
  atomicAddF(&den[(size_t)d * H + hh], pv);
}

// -------- edge pass 3: out[dst] += h[src] * alpha (one block per edge) --------
__global__ void edge_msg(const int* __restrict__ ei, int E, int EE, int H, int C,
                         int ldh, const float* __restrict__ h, const float* __restrict__ p,
                         const float* __restrict__ den, float* __restrict__ out, int ldo) {
  long long e = blockIdx.x;
  int tid = threadIdx.x;
  if (tid >= H * C) return;
  int s, d;
  if (e < E) { s = ei[e]; d = ei[(size_t)E + e]; } else { s = d = (int)(e - E); }
  int hh = tid / C;
  float alpha = p[(size_t)e * H + hh] / (den[(size_t)d * H + hh] + 1e-16f);
  float v = h[(size_t)s * ldh + tid] * alpha;
  atomicAddF(&out[(size_t)d * ldo + tid], v);
}

// ---------------- host launcher ----------------
extern "C" void kernel_launch(void* const* d_in, const int* in_sizes, int n_in,
                              void* d_out, int out_size, void* d_ws, size_t ws_size,
                              hipStream_t stream) {
  const float* x   = (const float*)d_in[0];
  const int*   ei  = (const int*)  d_in[1];
  const float* W1  = (const float*)d_in[2];
  const float* as1 = (const float*)d_in[3];
  const float* ad1 = (const float*)d_in[4];
  const float* b1  = (const float*)d_in[5];
  const float* W2  = (const float*)d_in[6];
  const float* as2 = (const float*)d_in[7];
  const float* ad2 = (const float*)d_in[8];
  const float* b2  = (const float*)d_in[9];

  const int N  = in_sizes[0] / INCH;
  const int E  = in_sizes[1] / 2;
  const int EE = E + N;
  const int Mtiles = (N + 15) / 16;
  const int KS1 = INCH / 32, NT1 = D1 / 16;     // 16, 16
  const int KS2 = D1 / 32,   NT2 = NCOL2 / 16;  // 8, 4

  // ---- workspace carve-up ----
  char* base = (char*)d_ws;
  size_t off = 0;
  auto carve = [&](size_t bytes) { void* p = base + off; off = (off + bytes + 255) & ~(size_t)255; return p; };

  const long long fragA1_elems = (long long)Mtiles * KS1 * 512;
  const int       fragB1_elems = NT1 * KS1 * 512;
  const long long fragA2_elems = (long long)Mtiles * KS2 * 512;
  const int       fragB2_elems = NT2 * KS2 * 512;

  __bf16* fragA1 = (__bf16*)carve((size_t)fragA1_elems * 2);  // reused as out1 (f32 N*D1)
  __bf16* fragB1 = (__bf16*)carve((size_t)fragB1_elems * 2);
  float*  h1     = (float*) carve((size_t)N * D1 * 4);
  float*  als1   = (float*) carve((size_t)N * HEADS1 * 4);
  float*  ald1   = (float*) carve((size_t)N * HEADS1 * 4);
  float*  m1     = (float*) carve((size_t)N * HEADS1 * 4);
  float*  den1   = (float*) carve((size_t)N * HEADS1 * 4);
  float*  p1     = (float*) carve((size_t)EE * HEADS1 * 4); // reused as fragA2 (bf16)
  __bf16* fragB2 = (__bf16*)carve((size_t)fragB2_elems * 2);
  float*  h2     = (float*) carve((size_t)N * NCOL2 * 4);
  float*  als2   = (float*) carve((size_t)N * 4);
  float*  ald2   = (float*) carve((size_t)N * 4);
  float*  m2     = (float*) carve((size_t)N * 4);
  float*  den2   = (float*) carve((size_t)N * 4);
  float*  p2     = (float*) carve((size_t)EE * 4);

  float*  out1   = (float*)fragA1;   // alias: fragA1 dead after gemm1
  __bf16* fragA2 = (__bf16*)p1;      // alias: p1 dead after edge_msg layer 1
  float*  outF   = (float*)d_out;

  const int TB = 256;
  auto blk = [&](long long n) { return (unsigned)((n + TB - 1) / TB); };

  // ===== Layer 1 =====
  conv_a_frag<<<blk(fragA1_elems), TB, 0, stream>>>(x, nullptr, 0, N, INCH, KS1, fragA1, fragA1_elems);
  conv_b_frag<<<blk(fragB1_elems), TB, 0, stream>>>(W1, INCH, D1, KS1, fragB1, fragB1_elems);

  // CT=8: each wave does 16 rows x 128 cols, A reused 8x from registers
  {
    int groups = Mtiles * (NT1 / 8);
    gemm_wmma_bf16<8><<<(groups + 7) / 8, 256, 0, stream>>>(fragA1, fragB1, h1, Mtiles, NT1, KS1, D1, N);
  }

  fill_f32<<<blk((long long)N * D1), TB, 0, stream>>>(out1, 0.f, (long long)N * D1);
  fill_f32<<<blk((long long)N * HEADS1), TB, 0, stream>>>(m1, -INFINITY, (long long)N * HEADS1);
  fill_f32<<<blk((long long)N * HEADS1), TB, 0, stream>>>(den1, 0.f, (long long)N * HEADS1);

  attn_logits8<<<blk((long long)N * HEADS1), TB, 0, stream>>>(h1, as1, ad1, als1, ald1, N);

  edge_pass1<<<blk((long long)EE * HEADS1), TB, 0, stream>>>(ei, E, EE, HEADS1, als1, ald1, p1, m1);
  edge_pass2<<<blk((long long)EE * HEADS1), TB, 0, stream>>>(ei, E, EE, HEADS1, m1, p1, den1);
  edge_msg<<<(unsigned)EE, 256, 0, stream>>>(ei, E, EE, HEADS1, HID1, D1, h1, p1, den1, out1, D1);

  // ===== Layer 2 =====
  conv_a_frag<<<blk(fragA2_elems), TB, 0, stream>>>(out1, b1, 1, N, D1, KS2, fragA2, fragA2_elems);
  conv_b_frag<<<blk(fragB2_elems), TB, 0, stream>>>(W2, D1, NCLS, KS2, fragB2, fragB2_elems);

  // CT=4: one wave covers the entire 64-col padded output per row tile
  {
    int groups = Mtiles * (NT2 / 4);
    gemm_wmma_bf16<4><<<(groups + 7) / 8, 256, 0, stream>>>(fragA2, fragB2, h2, Mtiles, NT2, KS2, NCOL2, N);
  }

  fill_f32<<<blk(N), TB, 0, stream>>>(m2, -INFINITY, N);
  fill_f32<<<blk(N), TB, 0, stream>>>(den2, 0.f, N);
  init_bias<<<blk((long long)N * NCLS), TB, 0, stream>>>(outF, b2, (long long)N * NCLS, NCLS);

  attn_logits1<<<blk(N), TB, 0, stream>>>(h2, as2, ad2, als2, ald2, N, NCLS, NCOL2);

  edge_pass1<<<blk(EE), TB, 0, stream>>>(ei, E, EE, 1, als2, ald2, p2, m2);
  edge_pass2<<<blk(EE), TB, 0, stream>>>(ei, E, EE, 1, m2, p2, den2);
  edge_msg<<<(unsigned)EE, 64, 0, stream>>>(ei, E, EE, 1, NCLS, NCOL2, h2, p2, den2, outF, NCLS);
}
